// AttentionModule_30099130810791
// MI455X (gfx1250) — matrix operand order repaired and checked by reading, not compiled, not run on hardware
//
#include <hip/hip_runtime.h>

#define B_ 4
#define S_ 4096
#define DM_ 512
#define DH_ 64
#define NEGV -1.0e9f

typedef __attribute__((ext_vector_type(16))) __bf16 v16bf;
typedef __attribute__((ext_vector_type(8)))  __bf16 v8bf;
typedef __attribute__((ext_vector_type(8)))  float  v8f;
typedef __attribute__((ext_vector_type(4)))  unsigned int v4u;
typedef __attribute__((ext_vector_type(8)))  int    v8i;
typedef __attribute__((ext_vector_type(4)))  int    v4i;

static __device__ inline v8f wmma_bf16(v16bf a, v16bf b, v8f c) {
  // D = A(16x32 bf16) * B(32x16 bf16) + C(16x16 f32)
  return __builtin_amdgcn_wmma_f32_16x16x32_bf16(false, a, false, b, (short)0, c,
                                                 false, false);
}

// Load a B-fragment: two 8-bf16 groups at p and p+16 (16B-aligned) per lane.
static __device__ inline v16bf load16(const __bf16* p) {
  v8bf g0 = *(const v8bf*)(p);
  v8bf g1 = *(const v8bf*)(p + 16);
  v16bf r;
#pragma unroll
  for (int i = 0; i < 8; ++i) { r[i] = g0[i]; r[i + 8] = g1[i]; }
  return r;
}

// ---------------------------------------------------------------------------
// Projection: [B*S, 512] (f32) x [512, 64] (f32) -> [B*S, 64] (bf16)
// blockIdx.y selects Q / K / V. 8 waves per block, each wave: 16x64 tile.
// ---------------------------------------------------------------------------
__global__ __launch_bounds__(256) void proj_qkv(
    const float* __restrict__ q,  const float* __restrict__ k,  const float* __restrict__ v,
    const float* __restrict__ Wq, const float* __restrict__ Wk, const float* __restrict__ Wv,
    __bf16* __restrict__ Qw, __bf16* __restrict__ Kw, __bf16* __restrict__ Vw) {
  __shared__ __align__(16) __bf16 Wt[DH_ * DM_];  // W transposed: [n][kdim], bf16

  const float* x; const float* W; __bf16* o;
  if (blockIdx.y == 0)      { x = q; W = Wq; o = Qw; }
  else if (blockIdx.y == 1) { x = k; W = Wk; o = Kw; }
  else                      { x = v; W = Wv; o = Vw; }

  int tid = threadIdx.x;
  for (int idx = tid; idx < DM_ * DH_; idx += 256) {
    int kk = idx / DH_, nn = idx % DH_;
    Wt[nn * DM_ + kk] = (__bf16)W[idx];
  }
  __syncthreads();

  int wave = tid >> 5, lane = tid & 31;
  int row0 = blockIdx.x * 128 + wave * 16;
  const float* xrow = x + (size_t)(row0 + (lane & 15)) * DM_;
  int aoff = (lane < 16) ? 0 : 8;   // A-layout K offset per lane half
  int bko  = (lane >> 4) << 3;      // B-layout K base per lane half

  v8f acc[4] = {};
  for (int k0 = 0; k0 < DM_; k0 += 32) {
    v16bf a;
    {
      const float* p0 = xrow + k0 + aoff;
      const float* p1 = p0 + 16;
#pragma unroll
      for (int i = 0; i < 8; ++i) { a[i] = (__bf16)p0[i]; a[i + 8] = (__bf16)p1[i]; }
    }
#pragma unroll
    for (int n = 0; n < 4; ++n) {
      const __bf16* bp = &Wt[(n * 16 + (lane & 15)) * DM_ + k0 + bko];
      acc[n] = wmma_bf16(a, load16(bp), acc[n]);
    }
  }

#pragma unroll
  for (int n = 0; n < 4; ++n)
#pragma unroll
    for (int r = 0; r < 8; ++r) {
      int row = row0 + r + ((lane >> 4) << 3);
      int col = n * 16 + (lane & 15);
      o[(size_t)row * DH_ + col] = (__bf16)acc[n][r];
    }
}

// ---------------------------------------------------------------------------
// Fused causal flash attention. Grid (S/128, B), 256 threads = 8 waves.
// Wave w owns query rows [i0, i0+16). K tile arrives via the Tensor Data
// Mover (async DMA, TENSORcnt); V tile is transposed into LDS manually.
// ---------------------------------------------------------------------------
__global__ __launch_bounds__(256) void flash_attn(
    const __bf16* __restrict__ Qw, const __bf16* __restrict__ Kw,
    const __bf16* __restrict__ Vw, float* __restrict__ out) {
  __shared__ __align__(16) __bf16 Kt[32 * DH_];     // [key][d] (TDM destination)
  __shared__ __align__(16) __bf16 Vt[DH_ * 32];     // [d][key] (transposed)
  __shared__ __align__(16) __bf16 Pst[8][16 * 32];  // per-wave P staging

  int tid = threadIdx.x, wave = tid >> 5, lane = tid & 31;
  int batch = blockIdx.y;
  int qblk  = blockIdx.x * 128;
  int i0    = qblk + wave * 16;

  const __bf16* Qp = Qw + (size_t)batch * S_ * DH_;
  const __bf16* Kp = Kw + (size_t)batch * S_ * DH_;
  const __bf16* Vp = Vw + (size_t)batch * S_ * DH_;

  int aoff = (lane < 16) ? 0 : 8;
  int bko  = (lane >> 4) << 3;

  // Q fragments (A-layout), head dim split into two K=32 chunks
  v16bf qa[2];
#pragma unroll
  for (int f = 0; f < 2; ++f) {
    const __bf16* p0 = Qp + (size_t)(i0 + (lane & 15)) * DH_ + f * 32 + aoff;
    v8bf g0 = *(const v8bf*)p0;
    v8bf g1 = *(const v8bf*)(p0 + 16);
#pragma unroll
    for (int i = 0; i < 8; ++i) { qa[f][i] = g0[i]; qa[f][i + 8] = g1[i]; }
  }

  v8f acc[4] = {};
  float m[8], l[8];
#pragma unroll
  for (int r = 0; r < 8; ++r) { m[r] = -1.0e30f; l[r] = 0.0f; }

  int ldk = tid >> 3;        // 0..31: key row
  int ldc = (tid & 7) * 8;   // 0..56: d column group
  int jmax = qblk + 127;     // causal extent of this block

  // LDS byte address of the K tile: flat shared addr truncated to 32 bits
  // (LDS aperture: low 32 bits give the LDS address).
  uint32_t klds = (uint32_t)(size_t)&Kt[0];

  for (int j0 = 0; j0 <= jmax; j0 += 32) {
    __syncthreads();

    if (wave == 0) {
      // ---- Tensor Data Mover: async 32x64 bf16 K tile -> LDS -------------
      uint64_t ga = (uint64_t)(size_t)(Kp + (size_t)j0 * DH_);
      v4u g0;
      g0[0] = 1u;                                          // count = 1 (valid D#)
      g0[1] = klds;                                        // lds_addr
      g0[2] = (uint32_t)ga;                                // global_addr[31:0]
      g0[3] = (uint32_t)((ga >> 32) & 0x01FFFFFFu)         // global_addr[56:32]
              | 0x80000000u;                               // type = 2 ("image")
      v8i g1;
      g1[0] = 0x00010000;                 // workgroup_mask=0, data_size=1 (2B)
      g1[1] = (int)(64u << 16);           // tensor_dim0 = 64 (row length d)
      g1[2] = (int)((unsigned)S_ << 16);  // tensor_dim1 = S keys
      g1[3] = (int)(64u << 16);           // tile_dim0 = 64
      g1[4] = 32;                         // tile_dim1 = 32 keys
      g1[5] = 64;                         // tensor_dim0_stride = 64
      g1[6] = 0;
      g1[7] = 0;
      v4i gz = {};
#if __has_include(<hip/amd_detail/amd_gfx1250_TDM.h>)
      v8i gz8 = {};
      __builtin_amdgcn_tensor_load_to_lds(g0, g1, gz, gz, gz8, 0);   // clang-23 form
#else
      __builtin_amdgcn_tensor_load_to_lds(g0, g1, gz, gz, 0);        // ROCm 7.2 form
#endif
    }

    {
      // V tile: global -> transposed LDS (TDM cannot transpose)
      v8bf vv = *(const v8bf*)(Vp + (size_t)(j0 + ldk) * DH_ + ldc);
#pragma unroll
      for (int i = 0; i < 8; ++i) Vt[(ldc + i) * 32 + ldk] = vv[i];
      if (j0 + 32 <= jmax) {  // global_prefetch_b8 next tiles
        __builtin_prefetch(Kp + (size_t)(j0 + 32 + ldk) * DH_ + ldc, 0, 1);
        __builtin_prefetch(Vp + (size_t)(j0 + 32 + ldk) * DH_ + ldc, 0, 1);
      }
    }

    if (wave == 0) __builtin_amdgcn_s_wait_tensorcnt(0);
    __syncthreads();
    if (j0 > i0 + 15) continue;  // fully masked for this wave; barriers kept uniform

    // S = Q K^T : two 16x16 score tiles, each two K=32 WMMAs
    v8f s[2] = {};
#pragma unroll
    for (int t = 0; t < 2; ++t)
#pragma unroll
      for (int f = 0; f < 2; ++f) {
        const __bf16* bp = &Kt[(t * 16 + (lane & 15)) * DH_ + f * 32 + bko];
        s[t] = wmma_bf16(qa[f], load16(bp), s[t]);
      }

    // scale (1/sqrt(64)) + causal mask (C layout: row = r + 8*(lane/16))
    int irow = i0 + ((lane >> 4) << 3);
#pragma unroll
    for (int t = 0; t < 2; ++t) {
      int j = j0 + t * 16 + (lane & 15);
#pragma unroll
      for (int r = 0; r < 8; ++r) {
        float val = s[t][r] * 0.125f;
        s[t][r] = (j <= irow + r) ? val : NEGV;
      }
    }

    // online softmax: row reductions across each 16-lane half
    float nm[8];
#pragma unroll
    for (int r = 0; r < 8; ++r) {
      float x = fmaxf(s[0][r], s[1][r]);
      x = fmaxf(x, __shfl_xor(x, 1, 32));
      x = fmaxf(x, __shfl_xor(x, 2, 32));
      x = fmaxf(x, __shfl_xor(x, 4, 32));
      x = fmaxf(x, __shfl_xor(x, 8, 32));
      nm[r] = fmaxf(m[r], x);
    }
#pragma unroll
    for (int t = 0; t < 2; ++t)
#pragma unroll
      for (int r = 0; r < 8; ++r)
        s[t][r] = __expf(s[t][r] - nm[r]);
#pragma unroll
    for (int r = 0; r < 8; ++r) {
      float x = s[0][r] + s[1][r];
      x += __shfl_xor(x, 1, 32);
      x += __shfl_xor(x, 2, 32);
      x += __shfl_xor(x, 4, 32);
      x += __shfl_xor(x, 8, 32);
      float cf = __expf(m[r] - nm[r]);
      l[r] = l[r] * cf + x;
      m[r] = nm[r];
#pragma unroll
      for (int n = 0; n < 4; ++n) acc[n][r] *= cf;
    }

    // P: C-layout -> LDS (row major) -> A-layout (same-wave DS, HW in-order)
    __bf16* pp = &Pst[wave][0];
#pragma unroll
    for (int t = 0; t < 2; ++t) {
      int col = t * 16 + (lane & 15);
#pragma unroll
      for (int r = 0; r < 8; ++r)
        pp[(r + ((lane >> 4) << 3)) * 32 + col] = (__bf16)s[t][r];
    }
    v16bf pa;
    {
      const __bf16* p0 = pp + (lane & 15) * 32 + aoff;
      v8bf g0 = *(const v8bf*)p0;
      v8bf g1 = *(const v8bf*)(p0 + 16);
#pragma unroll
      for (int i = 0; i < 8; ++i) { pa[i] = g0[i]; pa[i + 8] = g1[i]; }
    }

    // O += P V (Kdim = 32 keys)
#pragma unroll
    for (int n = 0; n < 4; ++n) {
      const __bf16* bp = &Vt[(n * 16 + (lane & 15)) * 32 + bko];
      acc[n] = wmma_bf16(pa, load16(bp), acc[n]);
    }
  }

  // epilogue: O / l
  float* op = out + (size_t)batch * S_ * DH_;
#pragma unroll
  for (int n = 0; n < 4; ++n)
#pragma unroll
    for (int r = 0; r < 8; ++r) {
      int row = i0 + r + ((lane >> 4) << 3);
      int col = n * 16 + (lane & 15);
      op[(size_t)row * DH_ + col] = acc[n][r] / l[r];
    }
}

extern "C" void kernel_launch(void* const* d_in, const int* in_sizes, int n_in,
                              void* d_out, int out_size, void* d_ws, size_t ws_size,
                              hipStream_t stream) {
  const float* q  = (const float*)d_in[0];
  const float* k  = (const float*)d_in[1];
  const float* v  = (const float*)d_in[2];
  const float* Wq = (const float*)d_in[3];
  const float* Wk = (const float*)d_in[4];
  const float* Wv = (const float*)d_in[5];
  // d_in[6] is the causal mask; causality is computed analytically in-kernel.

  size_t elems = (size_t)B_ * S_ * DH_;          // 1M elems per tensor
  __bf16* Qw = (__bf16*)d_ws;                    // 2 MB
  __bf16* Kw = Qw + elems;                       // 2 MB
  __bf16* Vw = Kw + elems;                       // 2 MB (total 6 MB of ws)

  dim3 pgrid((B_ * S_) / 128, 3);
  proj_qkv<<<pgrid, 256, 0, stream>>>(q, k, v, Wq, Wk, Wv, Qw, Kw, Vw);

  dim3 fgrid(S_ / 128, B_);
  flash_attn<<<fgrid, 256, 0, stream>>>(Qw, Kw, Vw, (float*)d_out);
}
